// MyMultiHeadAttention_66451734004229
// MI455X (gfx1250) — matrix operand order repaired and checked by reading, not compiled, not run on hardware
//
#include <hip/hip_runtime.h>
#include <hip/hip_bf16.h>

// ---------------------------------------------------------------------------
// MI455X (gfx1250, wave32) multi-head attention, WMMA bf16 path.
//
// Shapes: N=4 batches, LQ=LK=2048, D=1024, H=16 heads, HD=64.
// All matmuls use v_wmma_f32_16x16x32_bf16 (f32 accumulate).
// Roofline: the 1.07 GB attention-probability output dominates (46us @ 23.3TB/s
// just to write it); the 137 GFLOP of matmul is cheap on the WMMA pipe. So the
// attn tensor is written exactly once (coalesced, from LDS) and never re-read,
// and GEMM operands and V are staged through LDS (V with coalesced b128 loads).
// ---------------------------------------------------------------------------

typedef __attribute__((ext_vector_type(16))) __bf16 v16bf;
typedef __attribute__((ext_vector_type(8)))  float  v8f;

#define NB 4
#define LQ 2048
#define LK 2048
#define DM 1024
#define NH 16
#define HD 64

// A-matrix (16x32, 16-bit) element->K mapping (contiguous, matches B fills):
// lanes 0-15 hold K={0..15}, lanes 16-31 hold K={16..31},
// two K values per VGPR (element pair).
__device__ __forceinline__ int a_koff(int e, int lane) {
    return ((lane & 16) ? 16 : 0) + e;
}

__device__ __forceinline__ v8f wmma_bf16(v16bf a, v16bf b, v8f c) {
    return __builtin_amdgcn_wmma_f32_16x16x32_bf16(false, a, false, b, (short)0, c, false, false);
}

// ---------------------------------------------------------------------------
// Tiled GEMM: Y[M,N] = A[M,K] @ W[K,N] + bias[N]
// Block = 256 threads (8 waves). Macro tile 128(M) x 64(N); each wave owns a
// 32x32 tile (2x2 WMMA accumulators). A and W staged through LDS as bf16,
// with global_prefetch of the next k-tile overlapping the WMMA pipe.
// ---------------------------------------------------------------------------
template <bool A_BF16, bool OUT_BF16>
__global__ __launch_bounds__(256) void gemm_bf16_wmma(const void* __restrict__ Ap,
                                                      const float* __restrict__ W,
                                                      const float* __restrict__ bias,
                                                      void* __restrict__ Yp,
                                                      int M, int Nn, int K) {
    __shared__ __bf16 As[128][32];   // 8 KB
    __shared__ __bf16 Bt[64][32];    // 4 KB, W tile stored transposed [n][k]

    const int tid  = threadIdx.x;
    const int wv   = tid >> 5;
    const int lane = tid & 31;
    const int lh   = (lane & 16) ? 1 : 0;     // lane half
    const int l15  = lane & 15;
    const int wm   = wv >> 1;                 // 0..3  (M direction)
    const int wn   = wv & 1;                  // 0..1  (N direction)
    const int m0   = blockIdx.y * 128;
    const int n0   = blockIdx.x * 64;

    v8f acc[2][2];
    acc[0][0] = {}; acc[0][1] = {}; acc[1][0] = {}; acc[1][1] = {};

    for (int k0 = 0; k0 < K; k0 += 32) {
        // ---- prefetch next k-tile into cache (global_prefetch_b8) ----
        if (k0 + 32 < K) {
            int ra = tid >> 1;                          // 128 rows, 2 threads/row
            int ca = (tid & 1) * 16;
            if (A_BF16)
                __builtin_prefetch(&((const __bf16*)Ap)[(size_t)(m0 + ra) * K + k0 + 32 + ca], 0, 1);
            else
                __builtin_prefetch(&((const float*)Ap)[(size_t)(m0 + ra) * K + k0 + 32 + ca], 0, 1);
            int kw = tid >> 3;                          // 32 k-rows, 8 threads/row
            int nw = (tid & 7) * 8;
            __builtin_prefetch(&W[(size_t)(k0 + 32 + kw) * Nn + n0 + nw], 0, 1);
        }
        // ---- stage A tile (128x32) as bf16, coalesced along K ----
        #pragma unroll
        for (int i = 0; i < 16; ++i) {
            int idx = tid + i * 256;          // 0..4095
            int r = idx >> 5, c = idx & 31;
            float v;
            if (A_BF16) v = (float)((const __bf16*)Ap)[(size_t)(m0 + r) * K + k0 + c];
            else        v = ((const float*)Ap)[(size_t)(m0 + r) * K + k0 + c];
            As[r][c] = (__bf16)v;
        }
        // ---- stage W tile (32x64) transposed as bf16 ----
        #pragma unroll
        for (int i = 0; i < 8; ++i) {
            int idx = tid + i * 256;          // 0..2047
            int kk = idx >> 6, nn = idx & 63;
            Bt[nn][kk] = (__bf16)W[(size_t)(k0 + kk) * Nn + n0 + nn];
        }
        __syncthreads();

        // ---- load fragments from LDS in contiguous k order (lane half h: k = 16h+e) ----
        v16bf afr[2], bfr[2];
        #pragma unroll
        for (int t = 0; t < 2; ++t) {
            int row = 32 * wm + 16 * t + l15;
            #pragma unroll
            for (int e = 0; e < 16; ++e) afr[t][e] = As[row][a_koff(e, lane)];
        }
        #pragma unroll
        for (int u = 0; u < 2; ++u) {
            int col = 32 * wn + 16 * u + l15;
            #pragma unroll
            for (int e = 0; e < 16; ++e) bfr[u][e] = Bt[col][(lh ? 16 : 0) + e];
        }
        #pragma unroll
        for (int t = 0; t < 2; ++t)
            #pragma unroll
            for (int u = 0; u < 2; ++u)
                acc[t][u] = wmma_bf16(afr[t], bfr[u], acc[t][u]);
        __syncthreads();
    }

    // ---- epilogue: bias add, store (C/D layout: row = i + 8*laneHalf) ----
    #pragma unroll
    for (int t = 0; t < 2; ++t) {
        #pragma unroll
        for (int u = 0; u < 2; ++u) {
            int coll = 32 * wn + 16 * u + l15;
            float bv = bias[n0 + coll];
            #pragma unroll
            for (int i = 0; i < 8; ++i) {
                int rowl = 32 * wm + 16 * t + i + lh * 8;
                size_t off = (size_t)(m0 + rowl) * Nn + n0 + coll;
                float y = acc[t][u][i] + bv;
                if (OUT_BF16) ((__bf16*)Yp)[off] = (__bf16)y;
                else          ((float*)Yp)[off]  = y;
            }
        }
    }
}

// ---------------------------------------------------------------------------
// Fused attention: one block per (16 query rows, head, batch).
// Phase 1: S = QK^T * rsqrt(HD) + mask via WMMA, 8 waves x 256 keys -> LDS.
// Phase 2: block softmax over LDS (fully-masked rows -> 0, like nan_to_num),
//          stream normalized probs once to the attn output (coalesced).
// Phase 3: context = P @ V via WMMA; V chunks staged coalesced through LDS.
// ---------------------------------------------------------------------------
__global__ __launch_bounds__(256) void attn_fused(const __bf16* __restrict__ qp,
                                                  const __bf16* __restrict__ kp,
                                                  const __bf16* __restrict__ vp,
                                                  const int* __restrict__ mask,
                                                  float* __restrict__ attn_out,
                                                  __bf16* __restrict__ ao) {
    __shared__ float S[16][LK];      // 128 KB score strip (fits in 320 KB WGP LDS)
    __shared__ __bf16 Vs[32][HD];    // 4 KB staged V chunk
    __shared__ float red[16][16];
    __shared__ float rowmax[16];
    __shared__ float rowinv[16];

    const int tid  = threadIdx.x;
    const int wv   = tid >> 5;
    const int lane = tid & 31;
    const int lh   = (lane & 16) ? 1 : 0;
    const int l15  = lane & 15;
    const int q0   = blockIdx.x * 16;
    const int h    = blockIdx.y;
    const int n    = blockIdx.z;

    // ---- Phase 1: scores ----
    v16bf aq[2];
    {
        const __bf16* qrow = qp + ((size_t)n * LQ + q0 + l15) * DM + h * HD;
        #pragma unroll
        for (int t = 0; t < 2; ++t)
            #pragma unroll
            for (int e = 0; e < 16; ++e) aq[t][e] = qrow[32 * t + a_koff(e, lane)];
    }
    const int kv0 = wv * (LK / 8);   // 256 keys per wave
    for (int kt = 0; kt < 16; ++kt) {
        int j0 = kv0 + kt * 16;
        v16bf bk[2];
        const __bf16* krow = kp + ((size_t)n * LK + j0 + l15) * DM + h * HD;
        if (kt + 1 < 16)   // prefetch next key tile rows
            __builtin_prefetch(krow + 16 * DM, 0, 1);
        #pragma unroll
        for (int t = 0; t < 2; ++t)
            #pragma unroll
            for (int e = 0; e < 16; ++e) bk[t][e] = krow[32 * t + (lh ? 16 : 0) + e];
        v8f acc = {};
        acc = wmma_bf16(aq[0], bk[0], acc);
        acc = wmma_bf16(aq[1], bk[1], acc);
        int col = j0 + l15;
        bool pad = mask[n * LK + col] != 0;
        #pragma unroll
        for (int i = 0; i < 8; ++i) {
            float s = acc[i] * 0.125f;           // 1/sqrt(64)
            if (pad) s = -INFINITY;
            S[i + lh * 8][col] = s;
        }
    }
    __syncthreads();

    // ---- Phase 2: softmax (row = tid/16, 128-col segment per thread) ----
    const int row = tid >> 4, seg = tid & 15;
    float m = -INFINITY;
    #pragma unroll 4
    for (int c = 0; c < 128; ++c) m = fmaxf(m, S[row][seg * 128 + c]);
    red[row][seg] = m;
    __syncthreads();
    if (tid < 16) {
        float mm = -INFINITY;
        #pragma unroll
        for (int s2 = 0; s2 < 16; ++s2) mm = fmaxf(mm, red[tid][s2]);
        rowmax[tid] = mm;
    }
    __syncthreads();
    const float rm = rowmax[row];
    float sum = 0.f;
    #pragma unroll 4
    for (int c = 0; c < 128; ++c) {
        int cc = seg * 128 + c;
        float e2 = (rm == -INFINITY) ? 0.f : __expf(S[row][cc] - rm);
        S[row][cc] = e2;
        sum += e2;
    }
    red[row][seg] = sum;
    __syncthreads();
    if (tid < 16) {
        float t2 = 0.f;
        #pragma unroll
        for (int s2 = 0; s2 < 16; ++s2) t2 += red[tid][s2];
        rowinv[tid] = (t2 > 0.f) ? 1.f / t2 : 0.f;   // fully masked row -> 0
    }
    __syncthreads();
    {
        const float inv = rowinv[row];
        float* arow = attn_out + (((size_t)h * NB + n) * LQ + q0 + row) * LK;
        #pragma unroll 4
        for (int c = 0; c < 128; ++c) {
            int cc = seg * 128 + c;
            float p = S[row][cc] * inv;
            S[row][cc] = p;
            arow[cc] = p;                 // the single, coalesced 1 GB write
        }
    }
    __syncthreads();

    // ---- Phase 3: context = P @ V ----
    // Waves 0..3 each own a 16-wide HD column tile; all 8 waves cooperatively
    // stage each 32-key x 64-col V chunk into LDS with coalesced b128 loads.
    const int colb = (wv & 3) * 16;
    v8f acc = {};
    for (int kk = 0; kk < LK; kk += 32) {
        {
            int keyl = tid >> 3;               // 0..31 within chunk
            int cl   = (tid & 7) * 8;          // 8 contiguous bf16 = 16 bytes
            const __bf16* src = vp + ((size_t)n * LK + kk + keyl) * DM + h * HD + cl;
            *reinterpret_cast<uint4*>(&Vs[keyl][cl]) = *reinterpret_cast<const uint4*>(src);
        }
        __syncthreads();
        if (wv < 4) {
            v16bf ap, bvv;
            #pragma unroll
            for (int e = 0; e < 16; ++e) ap[e] = (__bf16)S[l15][kk + a_koff(e, lane)];
            #pragma unroll
            for (int e = 0; e < 16; ++e) bvv[e] = Vs[(lh ? 16 : 0) + e][colb + l15];
            acc = wmma_bf16(ap, bvv, acc);
        }
        __syncthreads();
    }
    if (wv < 4) {
        #pragma unroll
        for (int i = 0; i < 8; ++i) {
            int r = i + lh * 8;
            ao[((size_t)n * LQ + q0 + r) * DM + h * HD + colb + l15] = (__bf16)acc[i];
        }
    }
}

// ---------------------------------------------------------------------------
extern "C" void kernel_launch(void* const* d_in, const int* in_sizes, int n_in,
                              void* d_out, int out_size, void* d_ws, size_t ws_size,
                              hipStream_t stream) {
    (void)in_sizes; (void)n_in; (void)out_size; (void)ws_size;

    const float* query = (const float*)d_in[0];
    const float* key   = (const float*)d_in[1];
    const float* value = (const float*)d_in[2];
    const int*   kmask = (const int*)d_in[3];
    const float* Wq = (const float*)d_in[4];  const float* bq = (const float*)d_in[5];
    const float* Wk = (const float*)d_in[6];  const float* bk = (const float*)d_in[7];
    const float* Wv = (const float*)d_in[8];  const float* bv = (const float*)d_in[9];
    const float* Wo = (const float*)d_in[10]; const float* bo = (const float*)d_in[11];

    const size_t TOK = (size_t)NB * LQ;       // 8192 rows
    const size_t MAT = TOK * DM;              // 8.4M elements

    __bf16* qp = (__bf16*)d_ws;               // 16 MB each
    __bf16* kp = qp + MAT;
    __bf16* vp = kp + MAT;
    __bf16* ao = vp + MAT;                    // total 64 MB of d_ws

    float* out_f = (float*)d_out;             // [N,LQ,D]
    float* attn  = out_f + MAT;               // [H,N,LQ,LK]

    dim3 gblk(256);
    dim3 ggrid(DM / 64, (unsigned)(TOK / 128));

    // Q/K/V projections (f32 in -> bf16 out)
    gemm_bf16_wmma<false, true><<<ggrid, gblk, 0, stream>>>(query, Wq, bq, qp, (int)TOK, DM, DM);
    gemm_bf16_wmma<false, true><<<ggrid, gblk, 0, stream>>>(key,   Wk, bk, kp, (int)TOK, DM, DM);
    gemm_bf16_wmma<false, true><<<ggrid, gblk, 0, stream>>>(value, Wv, bv, vp, (int)TOK, DM, DM);

    // Fused attention: grid = (LQ/16 query strips, heads, batches)
    dim3 agrid(LQ / 16, NH, NB);
    attn_fused<<<agrid, gblk, 0, stream>>>(qp, kp, vp, kmask, attn, ao);

    // Output projection (bf16 in -> f32 out)
    gemm_bf16_wmma<true, false><<<ggrid, gblk, 0, stream>>>(ao, Wo, bo, out_f, (int)TOK, DM, DM);
}
